// MultiheadAttention_90984587198868
// MI455X (gfx1250) — matrix operand order repaired and checked by reading, not compiled, not run on hardware
//
#include <hip/hip_runtime.h>
#include <hip/hip_bf16.h>
#include <stdint.h>

// ---------------------------------------------------------------------------
// Complex MHA for MI455X (gfx1250): all GEMMs via v_wmma_f32_16x16x32_bf16.
//   E=512 H=8 D=64 T=S=2048 B=4
// GEMMs register-blocked 32x32 per wave (4 WMMA / k-step). Both projection
// GEMMs stage their B slab through LDS with the Tensor Data Mover
// (tensor_load_to_lds + s_wait_tensorcnt), double-buffered so the DMA
// overlaps WMMA compute; A streams via clause'd global_load_b128 + prefetch.
// ---------------------------------------------------------------------------
#define E_DIM   512
#define H_HEADS 8
#define D_HEAD  64
#define T_LEN   2048
#define S_LEN   2048
#define B_BATCH 4
#define ROWS    (T_LEN * B_BATCH)   // 8192
#define KCAT    (2 * E_DIM)         // 1024 : [real | imag] along K
#define NCAT    (2 * E_DIM)         // 1024 : [yr | yi] along N

typedef __bf16 bf16_t;
typedef __attribute__((ext_vector_type(16))) __bf16 v16bf;
typedef __attribute__((ext_vector_type(8)))  float  v8f;
typedef unsigned int u32x4 __attribute__((ext_vector_type(4)));
typedef int          i32x4 __attribute__((ext_vector_type(4)));
typedef int          i32x8 __attribute__((ext_vector_type(8)));

union FragU { v16bf v; uint4 q[2]; };

__device__ __forceinline__ bf16_t f2bf(float f) {
    union { float f; uint32_t u; } c; c.f = f;
    uint32_t r = c.u + 0x7FFFu + ((c.u >> 16) & 1u);   // round-to-nearest-even
    union { uint16_t s; bf16_t b; } o; o.s = (uint16_t)(r >> 16);
    return o.b;
}

// A-matrix fragment (16x32 bf16): per lane, elements 0..7 = K {0..7|16..23},
// elements 8..15 = K {8..15|24..31}  -> two contiguous 16B loads.
__device__ __forceinline__ v16bf load_a_frag(const bf16_t* p0, const bf16_t* p1) {
    FragU f;
    f.q[0] = *reinterpret_cast<const uint4*>(p0);
    f.q[1] = *reinterpret_cast<const uint4*>(p1);
    return f.v;
}
// B-matrix fragment (32x16 bf16): per lane (column = lane%16), 16 contiguous
// K values (lane<16: K 0..15, lane>=16: K 16..31) -> one contiguous 32B load.
__device__ __forceinline__ v16bf load_b_frag(const bf16_t* p) {
    FragU f;
    const uint4* q = reinterpret_cast<const uint4*>(p);
    f.q[0] = q[0]; f.q[1] = q[1];
    return f.v;
}
__device__ __forceinline__ v8f wmma_bf16(v16bf a, v16bf b, v8f c) {
    return __builtin_amdgcn_wmma_f32_16x16x32_bf16(false, a, false, b,
                                                   (short)0, c, false, false);
}

// ---------------------------------------------------------------------------
// Tensor Data Mover: 2D tile (tile_k contiguous elems x tile_rows rows) of
// bf16 from global -> LDS. D# packed per CDNA5 ISA 8.3/8.4:
//   group0: count=1 | lds_addr | global_addr(57b) | type=2
//   group1: data_size=1(2B); tensor_dim0/1; tile_dim0/1; tensor_dim0_stride
// groups 2/3 zero (2D tensor).
// ---------------------------------------------------------------------------
__device__ __forceinline__ void tdm_load_2d_bf16(unsigned int lds_byte_addr,
                                                 const void* gptr,
                                                 unsigned int tile_k,
                                                 unsigned int tile_rows,
                                                 unsigned int row_stride) {
    unsigned long long ga = (unsigned long long)(uintptr_t)gptr;
    u32x4 g0;
    g0[0] = 0x1u;                                          // count=1
    g0[1] = lds_byte_addr;                                 // lds_addr
    g0[2] = (unsigned int)(ga & 0xFFFFFFFFu);              // global_addr[31:0]
    g0[3] = (unsigned int)((ga >> 32) & 0x01FFFFFFu)       // global_addr[56:32]
            | (2u << 30);                                  // type=2 (image)
    const unsigned int dim0 = row_stride;                  // >= tile_k
    const unsigned int dim1 = 1u << 20;                    // >= tile_rows
    i32x8 g1;
    g1[0] = (int)(1u << 16);                               // data_size=1 (2B)
    g1[1] = (int)((dim0 & 0xFFFFu) << 16);                 // tensor_dim0[15:0]
    g1[2] = (int)((dim0 >> 16) | ((dim1 & 0xFFFFu) << 16));
    g1[3] = (int)((dim1 >> 16) | (tile_k << 16));          // tile_dim0
    g1[4] = (int)tile_rows;                                // tile_dim1 (dim2=0)
    g1[5] = (int)row_stride;                               // stride0[31:0]
    g1[6] = 0;                                             // stride0[47:32]
    g1[7] = 0;
    i32x4 gz = {0, 0, 0, 0};
#if defined(__clang_major__) && (__clang_major__ >= 23)
    i32x8 gz8 = {0, 0, 0, 0, 0, 0, 0, 0};
    __builtin_amdgcn_tensor_load_to_lds(g0, g1, gz, gz, gz8, 0);
#else
    __builtin_amdgcn_tensor_load_to_lds(g0, g1, gz, gz, 0);
#endif
}

// ---------------------------------------------------------------------------
// Prep kernels: build bf16 concatenated operands.
// ---------------------------------------------------------------------------
__global__ void build_xcat(const float* __restrict__ qr, const float* __restrict__ qi,
                           const float* __restrict__ kr, const float* __restrict__ ki,
                           const float* __restrict__ vr, const float* __restrict__ vi,
                           bf16_t* __restrict__ xcat) {
    const long total = 3L * ROWS * KCAT;
    for (long i = (long)blockIdx.x * blockDim.x + threadIdx.x; i < total;
         i += (long)gridDim.x * blockDim.x) {
        long z   = i / ((long)ROWS * KCAT);
        long rem = i - z * (long)ROWS * KCAT;
        long row = rem >> 10;            // (t*B + b)
        long c   = rem & 1023;
        const float* src;
        if      (z == 0) src = (c < E_DIM) ? qr : qi;
        else if (z == 1) src = (c < E_DIM) ? kr : ki;
        else             src = (c < E_DIM) ? vr : vi;
        xcat[i] = f2bf(src[row * E_DIM + (c & (E_DIM - 1))]);
    }
}

__global__ void build_wcat_in(const float* __restrict__ iwr, const float* __restrict__ iwi,
                              bf16_t* __restrict__ wcat) {
    const long total = 3L * NCAT * KCAT;
    for (long i = (long)blockIdx.x * blockDim.x + threadIdx.x; i < total;
         i += (long)gridDim.x * blockDim.x) {
        long z   = i / ((long)NCAT * KCAT);
        long rem = i - z * (long)NCAT * KCAT;
        long n   = rem >> 10;
        long k   = rem & 1023;
        int part = (int)(n >> 9), nc = (int)(n & 511);
        int kh   = (int)(k >> 9), kc = (int)(k & 511);
        long wrow = ((long)z * E_DIM + nc) * E_DIM + kc;
        float v;
        if (part == 0) v = (kh == 0) ? iwr[wrow] : -iwi[wrow];
        else           v = (kh == 0) ? iwi[wrow] :  iwr[wrow];
        wcat[i] = f2bf(v);
    }
}

__global__ void build_wcat_out(const float* __restrict__ owr, const float* __restrict__ owi,
                               bf16_t* __restrict__ wcat) {
    const long total = (long)NCAT * KCAT;
    for (long i = (long)blockIdx.x * blockDim.x + threadIdx.x; i < total;
         i += (long)gridDim.x * blockDim.x) {
        long n = i >> 10, k = i & 1023;
        int part = (int)(n >> 9), nc = (int)(n & 511);
        int kh   = (int)(k >> 9), kc = (int)(k & 511);
        long wrow = (long)nc * E_DIM + kc;
        float v;
        if (part == 0) v = (kh == 0) ? owr[wrow] : -owi[wrow];
        else           v = (kh == 0) ? owi[wrow] :  owr[wrow];
        wcat[i] = f2bf(v);
    }
}

// ---------------------------------------------------------------------------
// In-projection GEMM (per z in {q,k,v}): (8192x1024)@(1024x1024) bf16->f32.
// Block = 8 waves (4 along M, 2 along N); each wave: 32x32 C region = 4 tiles.
// B slab (64 rows x 128 k) TDM-staged into LDS, double-buffered.
// Epilogue scatters: Q,K -> [B,H,T,D] ; V -> [B,H,D,S] (transposed).
// ---------------------------------------------------------------------------
__global__ void __launch_bounds__(256)
gemm_inproj(const bf16_t* __restrict__ xcat, const bf16_t* __restrict__ wcat,
            const float* __restrict__ in_b_r, const float* __restrict__ in_b_i,
            bf16_t* __restrict__ q_r, bf16_t* __restrict__ q_i,
            bf16_t* __restrict__ k_r, bf16_t* __restrict__ k_i,
            bf16_t* __restrict__ vT_r, bf16_t* __restrict__ vT_i) {
    __shared__ __align__(16) bf16_t s_bslab[2][64][128];   // 32 KB
    const int z  = blockIdx.z;
    const bf16_t* A  = xcat + (size_t)z * ROWS * KCAT;
    const bf16_t* Bw = wcat + (size_t)z * NCAT * KCAT;
    const int tid = threadIdx.x, lane = tid & 31, wv = tid >> 5;
    const int ln16 = lane & 15, lhalf = lane >> 4;
    const int mq = wv & 3, nq = wv >> 2;                   // wave grid 4x2
    const int m0 = blockIdx.x * 128 + mq * 32;
    const int n0b = blockIdx.y * 64;                       // block's B slab rows
    const int nq32 = nq * 32;                              // wave offset in slab
    const bf16_t* arow0 = A + (size_t)(m0 + ln16) * KCAT;
    const bf16_t* arow1 = A + (size_t)(m0 + 16 + ln16) * KCAT;

    if (wv == 0) {
        tdm_load_2d_bf16((unsigned int)(uintptr_t)&s_bslab[0][0][0],
                         Bw + (size_t)n0b * KCAT, 128, 64, KCAT);
        __builtin_amdgcn_s_wait_tensorcnt(0);
    }
    __syncthreads();

    v8f acc00 = {}, acc01 = {}, acc10 = {}, acc11 = {};
    const int NCHUNK = KCAT / 128;                         // 8
    for (int chunk = 0; chunk < NCHUNK; ++chunk) {
        const int cur = chunk & 1;
        if (wv == 0 && chunk + 1 < NCHUNK) {
            tdm_load_2d_bf16((unsigned int)(uintptr_t)&s_bslab[cur ^ 1][0][0],
                             Bw + (size_t)n0b * KCAT + (chunk + 1) * 128,
                             128, 64, KCAT);
        }
        const int kg = chunk * 128;
#pragma unroll
        for (int ks = 0; ks < 128; ks += 32) {
            const int kk = kg + ks;
            __builtin_prefetch(arow0 + kk + 256, 0, 1);    // global_prefetch_b8
            v16bf a0 = load_a_frag(arow0 + kk + lhalf * 16, arow0 + kk + 8 + lhalf * 16);
            v16bf a1 = load_a_frag(arow1 + kk + lhalf * 16, arow1 + kk + 8 + lhalf * 16);
            v16bf b0 = load_b_frag(&s_bslab[cur][nq32 + ln16][ks + lhalf * 16]);
            v16bf b1 = load_b_frag(&s_bslab[cur][nq32 + 16 + ln16][ks + lhalf * 16]);
            acc00 = wmma_bf16(a0, b0, acc00);
            acc01 = wmma_bf16(a0, b1, acc01);
            acc10 = wmma_bf16(a1, b0, acc10);
            acc11 = wmma_bf16(a1, b1, acc11);
        }
        if (wv == 0) __builtin_amdgcn_s_wait_tensorcnt(0);
        __syncthreads();
    }

    v8f accs[2][2] = {{acc00, acc01}, {acc10, acc11}};
#pragma unroll
    for (int mi = 0; mi < 2; ++mi)
#pragma unroll
        for (int ni = 0; ni < 2; ++ni)
#pragma unroll
            for (int r = 0; r < 8; ++r) {
                int grow = m0 + mi * 16 + r + 8 * lhalf;   // C/D: row = r + 8*(lane/16)
                int gcol = n0b + nq32 + ni * 16 + ln16;
                int t    = grow >> 2, b = grow & 3;        // rows are (t*B + b)
                int part = gcol >> 9, nc = gcol & 511;
                int h    = nc >> 6,   d  = nc & 63;
                float bias = (part ? in_b_i : in_b_r)[z * E_DIM + nc];
                bf16_t val = f2bf(accs[mi][ni][r] + bias);
                if (z == 0)
                    (part ? q_i : q_r)[(((size_t)(b * H_HEADS + h)) * T_LEN + t) * D_HEAD + d] = val;
                else if (z == 1)
                    (part ? k_i : k_r)[(((size_t)(b * H_HEADS + h)) * S_LEN + t) * D_HEAD + d] = val;
                else
                    (part ? vT_i : vT_r)[(((size_t)(b * H_HEADS + h)) * D_HEAD + d) * S_LEN + t] = val;
            }
}

// ---------------------------------------------------------------------------
// Attention: one block per (b, 16-row t-tile); loop over 8 heads.
// LDS: 128 KB scores (f32) + 64 KB P (bf16) + 1 KB reduce  (WGP has 320 KB).
// K/V fragments are wave-distinct (no cross-wave redundancy), so they stream
// directly from global/L2.
// ---------------------------------------------------------------------------
__global__ void __launch_bounds__(256)
attn_kernel(const bf16_t* __restrict__ q_r, const bf16_t* __restrict__ q_i,
            const bf16_t* __restrict__ k_r, const bf16_t* __restrict__ k_i,
            const bf16_t* __restrict__ vT_r, const bf16_t* __restrict__ vT_i,
            bf16_t* __restrict__ ocat, float* __restrict__ attn) {
    __shared__ __align__(16) float  s_scores[16][S_LEN];
    __shared__ __align__(16) bf16_t s_p[16][S_LEN];
    __shared__ float s_red[16][16];

    const int tid  = threadIdx.x;
    const int lane = tid & 31, wv = tid >> 5;
    const int ln16 = lane & 15, lhalf = lane >> 4;
    const int b  = blockIdx.y;
    const int t0 = blockIdx.x * 16;
    const int srow = tid >> 4, scol = tid & 15;          // softmax mapping

    for (int h = 0; h < H_HEADS; ++h) {
        const size_t headQ = ((size_t)(b * H_HEADS + h)) * T_LEN;
        const size_t headK = ((size_t)(b * H_HEADS + h)) * S_LEN;

        // Q fragments (A-matrix, identical across waves): rows t0..t0+15, K=D=64
        const bf16_t* qrow_r = q_r + (headQ + t0 + ln16) * D_HEAD;
        const bf16_t* qrow_i = q_i + (headQ + t0 + ln16) * D_HEAD;
        const int a0 = lhalf * 16;
        v16bf qr0 = load_a_frag(qrow_r + a0,      qrow_r + a0 + 8);
        v16bf qr1 = load_a_frag(qrow_r + 32 + a0, qrow_r + 32 + a0 + 8);
        v16bf qi0 = load_a_frag(qrow_i + a0,      qrow_i + a0 + 8);
        v16bf qi1 = load_a_frag(qrow_i + 32 + a0, qrow_i + 32 + a0 + 8);

        // ---- scores: each wave owns every 8th 16-wide s-tile ----
        for (int st = wv; st < S_LEN / 16; st += 8) {
            const int s0 = st * 16;
            const bf16_t* krow_r = k_r + (headK + s0 + ln16) * D_HEAD + lhalf * 16;
            const bf16_t* krow_i = k_i + (headK + s0 + ln16) * D_HEAD + lhalf * 16;
            v8f acc = {};
            acc = wmma_bf16(qr0, load_b_frag(krow_r),      acc);
            acc = wmma_bf16(qr1, load_b_frag(krow_r + 32), acc);
            acc = wmma_bf16(qi0, load_b_frag(krow_i),      acc);
            acc = wmma_bf16(qi1, load_b_frag(krow_i + 32), acc);
#pragma unroll
            for (int r = 0; r < 8; ++r)
                s_scores[r + 8 * lhalf][s0 + ln16] = acc[r] * 0.125f;  // 1/sqrt(64)
        }
        __syncthreads();

        // ---- softmax over S (16 threads per row, coalesced stride-16) ----
        float lmax = -3.0e38f;
        for (int j = 0; j < S_LEN / 16; ++j)
            lmax = fmaxf(lmax, s_scores[srow][scol + 16 * j]);
        s_red[srow][scol] = lmax;
        __syncthreads();
        float rmax = -3.0e38f;
#pragma unroll
        for (int t = 0; t < 16; ++t) rmax = fmaxf(rmax, s_red[srow][t]);
        __syncthreads();

        float lsum = 0.f;
        for (int j = 0; j < S_LEN / 16; ++j) {
            int s = scol + 16 * j;
            float e = __expf(s_scores[srow][s] - rmax);
            s_scores[srow][s] = e;
            lsum += e;
        }
        s_red[srow][scol] = lsum;
        __syncthreads();
        float rsum = 0.f;
#pragma unroll
        for (int t = 0; t < 16; ++t) rsum += s_red[srow][t];
        const float rinv = 1.0f / rsum;

        float* arow_g = attn + ((size_t)(b * T_LEN + t0 + srow)) * S_LEN;
        for (int j = 0; j < S_LEN / 16; ++j) {
            int s = scol + 16 * j;
            float pn = s_scores[srow][s] * rinv;
            s_p[srow][s] = f2bf(pn);                      // normalized P (bf16)
            float contrib = 0.125f * pn;                  // mean over H=8 heads
            if (h == 0) arow_g[s] = contrib; else arow_g[s] += contrib;
        }
        __syncthreads();

        // ---- O = P @ V : wave -> (part = r|i, 16-wide d-tile) ----
        const int part = wv >> 2, dt = wv & 3;
        const bf16_t* vT   = part ? vT_i : vT_r;
        const bf16_t* vrow = vT + (((size_t)(b * H_HEADS + h)) * D_HEAD +
                                   dt * 16 + ln16) * S_LEN + lhalf * 16;
        v8f acc = {};
        for (int sk = 0; sk < S_LEN; sk += 32) {
            v16bf a  = load_a_frag(&s_p[ln16][sk + lhalf * 16],
                                   &s_p[ln16][sk + 8 + lhalf * 16]);
            v16bf bm = load_b_frag(vrow + sk);
            acc = wmma_bf16(a, bm, acc);
        }
#pragma unroll
        for (int r = 0; r < 8; ++r) {
            int m = r + 8 * lhalf;
            size_t row = (size_t)(t0 + m) * B_BATCH + b;
            ocat[row * KCAT + part * E_DIM + h * D_HEAD + dt * 16 + ln16] = f2bf(acc[r]);
        }
        __syncthreads();
    }
}

// ---------------------------------------------------------------------------
// Out-projection GEMM: (8192x1024)@(1024x1024) -> f32 out_r|out_i in d_out.
// B slab TDM-staged into LDS, double-buffered (same scheme as gemm_inproj).
// ---------------------------------------------------------------------------
__global__ void __launch_bounds__(256)
gemm_outproj(const bf16_t* __restrict__ ocat, const bf16_t* __restrict__ wocat,
             const float* __restrict__ ob_r, const float* __restrict__ ob_i,
             float* __restrict__ out) {
    __shared__ __align__(16) bf16_t s_bslab[2][64][128];   // 32 KB
    const int tid = threadIdx.x, lane = tid & 31, wv = tid >> 5;
    const int ln16 = lane & 15, lhalf = lane >> 4;
    const int mq = wv & 3, nq = wv >> 2;
    const int m0 = blockIdx.x * 128 + mq * 32;
    const int n0b = blockIdx.y * 64;                       // block's B slab rows
    const int nq32 = nq * 32;                              // wave offset in slab
    const bf16_t* arow0 = ocat + (size_t)(m0 + ln16) * KCAT;
    const bf16_t* arow1 = ocat + (size_t)(m0 + 16 + ln16) * KCAT;

    if (wv == 0) {
        tdm_load_2d_bf16((unsigned int)(uintptr_t)&s_bslab[0][0][0],
                         wocat + (size_t)n0b * KCAT, 128, 64, KCAT);
        __builtin_amdgcn_s_wait_tensorcnt(0);
    }
    __syncthreads();

    v8f acc00 = {}, acc01 = {}, acc10 = {}, acc11 = {};
    const int NCHUNK = KCAT / 128;                         // 8
    for (int chunk = 0; chunk < NCHUNK; ++chunk) {
        const int cur = chunk & 1;
        if (wv == 0 && chunk + 1 < NCHUNK) {
            tdm_load_2d_bf16((unsigned int)(uintptr_t)&s_bslab[cur ^ 1][0][0],
                             wocat + (size_t)n0b * KCAT + (chunk + 1) * 128,
                             128, 64, KCAT);
        }
        const int kg = chunk * 128;
#pragma unroll
        for (int ks = 0; ks < 128; ks += 32) {
            const int kk = kg + ks;
            __builtin_prefetch(arow0 + kk + 256, 0, 1);
            v16bf a0 = load_a_frag(arow0 + kk + lhalf * 16, arow0 + kk + 8 + lhalf * 16);
            v16bf a1 = load_a_frag(arow1 + kk + lhalf * 16, arow1 + kk + 8 + lhalf * 16);
            v16bf b0 = load_b_frag(&s_bslab[cur][nq32 + ln16][ks + lhalf * 16]);
            v16bf b1 = load_b_frag(&s_bslab[cur][nq32 + 16 + ln16][ks + lhalf * 16]);
            acc00 = wmma_bf16(a0, b0, acc00);
            acc01 = wmma_bf16(a0, b1, acc01);
            acc10 = wmma_bf16(a1, b0, acc10);
            acc11 = wmma_bf16(a1, b1, acc11);
        }
        if (wv == 0) __builtin_amdgcn_s_wait_tensorcnt(0);
        __syncthreads();
    }

    v8f accs[2][2] = {{acc00, acc01}, {acc10, acc11}};
#pragma unroll
    for (int mi = 0; mi < 2; ++mi)
#pragma unroll
        for (int ni = 0; ni < 2; ++ni)
#pragma unroll
            for (int r = 0; r < 8; ++r) {
                int grow = m0 + mi * 16 + r + 8 * lhalf;
                int gcol = n0b + nq32 + ni * 16 + ln16;
                int part = gcol >> 9, nc = gcol & 511;
                float v  = accs[mi][ni][r] + (part ? ob_i : ob_r)[nc];
                out[(size_t)part * ((size_t)ROWS * E_DIM) + (size_t)grow * E_DIM + nc] = v;
            }
}

// ---------------------------------------------------------------------------
extern "C" void kernel_launch(void* const* d_in, const int* in_sizes, int n_in,
                              void* d_out, int out_size, void* d_ws, size_t ws_size,
                              hipStream_t stream) {
    const float* query_r = (const float*)d_in[0];
    const float* query_i = (const float*)d_in[1];
    const float* key_r   = (const float*)d_in[2];
    const float* key_i   = (const float*)d_in[3];
    const float* value_r = (const float*)d_in[4];
    const float* value_i = (const float*)d_in[5];
    const float* in_w_r  = (const float*)d_in[6];
    const float* in_w_i  = (const float*)d_in[7];
    const float* in_b_r  = (const float*)d_in[8];
    const float* in_b_i  = (const float*)d_in[9];
    const float* out_w_r = (const float*)d_in[10];
    const float* out_w_i = (const float*)d_in[11];
    const float* out_b_r = (const float*)d_in[12];
    const float* out_b_i = (const float*)d_in[13];

    char* ws = (char*)d_ws;
    size_t off = 0;
    auto alloc = [&](size_t bytes) -> void* {
        off = (off + 255) & ~(size_t)255;
        void* p = ws + off;
        off += bytes;
        return p;
    };
    bf16_t* xcat     = (bf16_t*)alloc(3ull * ROWS * KCAT * sizeof(bf16_t));
    bf16_t* wcat_in  = (bf16_t*)alloc(3ull * NCAT * KCAT * sizeof(bf16_t));
    bf16_t* wcat_out = (bf16_t*)alloc((size_t)NCAT * KCAT * sizeof(bf16_t));
    const size_t headElems = (size_t)B_BATCH * H_HEADS * T_LEN * D_HEAD;  // 4.19M
    bf16_t* q_r  = (bf16_t*)alloc(headElems * sizeof(bf16_t));
    bf16_t* q_i  = (bf16_t*)alloc(headElems * sizeof(bf16_t));
    bf16_t* k_r  = (bf16_t*)alloc(headElems * sizeof(bf16_t));
    bf16_t* k_i  = (bf16_t*)alloc(headElems * sizeof(bf16_t));
    bf16_t* vT_r = (bf16_t*)alloc(headElems * sizeof(bf16_t));
    bf16_t* vT_i = (bf16_t*)alloc(headElems * sizeof(bf16_t));
    bf16_t* ocat = (bf16_t*)alloc((size_t)ROWS * KCAT * sizeof(bf16_t));

    build_xcat<<<8192, 256, 0, stream>>>(query_r, query_i, key_r, key_i,
                                         value_r, value_i, xcat);
    build_wcat_in<<<2048, 256, 0, stream>>>(in_w_r, in_w_i, wcat_in);
    build_wcat_out<<<1024, 256, 0, stream>>>(out_w_r, out_w_i, wcat_out);

    gemm_inproj<<<dim3(ROWS / 128, NCAT / 64, 3), 256, 0, stream>>>(
        xcat, wcat_in, in_b_r, in_b_i, q_r, q_i, k_r, k_i, vT_r, vT_i);

    float* attn = (float*)d_out + 2ull * ROWS * E_DIM;   // after out_r|out_i
    attn_kernel<<<dim3(T_LEN / 16, B_BATCH), 256, 0, stream>>>(
        q_r, q_i, k_r, k_i, vT_r, vT_i, ocat, attn);

    gemm_outproj<<<dim3(ROWS / 128, NCAT / 64), 256, 0, stream>>>(
        ocat, wcat_out, out_b_r, out_b_i, (float*)d_out);
}
